// BertEncoder_52922587021630
// MI455X (gfx1250) — matrix operand order repaired and checked by reading, not compiled, not run on hardware
//
#include <hip/hip_runtime.h>
#include <hip/hip_bf16.h>

// ---------------------------------------------------------------------------
// BERT encoder (L=6, B=16, S=512, H=768, NH=12, DH=64, I=3072) for gfx1250.
// All matmuls run on v_wmma_f32_16x16x32_bf16 (bf16 inputs, f32 accumulate).
// Weights are converted fp32->bf16 AND transposed (N-major) once per layer so
// GEMM staging is fully vectorized (b128 global loads + b128 LDS stores).
// Per-wave output arrangement is 2 M-tiles x 4 N-tiles: each B fragment feeds
// two independent WMMAs, halving LDS-wait stall points and cutting ds traffic.
// ---------------------------------------------------------------------------

typedef __attribute__((ext_vector_type(16))) __bf16 v16bf;
typedef __attribute__((ext_vector_type(8)))  __bf16 v8bf;
typedef __attribute__((ext_vector_type(8)))  float  v8f;

union Frag16 { v16bf v; v8bf h[2]; };

// Load 8 contiguous fp32 (16B-aligned) and convert to 8 bf16.
__device__ __forceinline__ v8bf cvt8(const float* __restrict__ p) {
    float4 x = *(const float4*)p;
    float4 y = *(const float4*)(p + 4);
    v8bf r;
    r[0] = (__bf16)x.x; r[1] = (__bf16)x.y; r[2] = (__bf16)x.z; r[3] = (__bf16)x.w;
    r[4] = (__bf16)y.x; r[5] = (__bf16)y.y; r[6] = (__bf16)y.z; r[7] = (__bf16)y.w;
    return r;
}

// ---------------------------------------------------------------------------
// Fused fp32 -> bf16 convert + transpose:  dst[n*K + k] = (bf16)src[k*N + n]
// Tiled 32x32 through LDS; 256 threads (32x8), grid (N/32, K/32).
// ---------------------------------------------------------------------------
__global__ __launch_bounds__(256)
void convert_transpose_kernel(const float* __restrict__ src, __bf16* __restrict__ dst,
                              int K, int N)
{
    __shared__ float tile[32][33];
    const int kb = blockIdx.y * 32, nb = blockIdx.x * 32;
    const int tx = threadIdx.x & 31, ty = threadIdx.x >> 5;   // ty: 0..7
#pragma unroll
    for (int i = 0; i < 4; ++i) {
        int kk = ty + i * 8;
        tile[kk][tx] = src[(size_t)(kb + kk) * N + nb + tx];
    }
    __syncthreads();
#pragma unroll
    for (int i = 0; i < 4; ++i) {
        int nn = ty + i * 8;
        dst[(size_t)(nb + nn) * K + kb + tx] = (__bf16)tile[tx][nn];
    }
}

// ---------------------------------------------------------------------------
// Tiled WMMA GEMM:  C[M,N] = act( A[M,K] @ W[K,N] + bias[N] (+ res[M,N]) )
// A fp32, WT = bf16 W transposed (N x K), C fp32.
// 256 threads (8 waves), 128x128 tile, K-step 64 with register prefetch.
// LDS rows padded to 72 bf16 (144 B): conflict-free, 16B-aligned b128 frags.
// Wave (wm = wave>>1, wn = wave&1) owns M-tiles {wm, wm+4}, N-tiles 4wn..4wn+3.
// act: 0 = none, 1 = exact GELU.
// ---------------------------------------------------------------------------
#define TKP 72

__global__ __launch_bounds__(256)
void gemm_bias_kernel(const float* __restrict__ A, const __bf16* __restrict__ WT,
                      const float* __restrict__ bias, const float* __restrict__ res,
                      float* __restrict__ C, int M, int N, int K, int act)
{
    __shared__ __align__(16) __bf16 As[128][TKP];   // 18 KB (M x K-step)
    __shared__ __align__(16) __bf16 Bs[128][TKP];   // 18 KB (N x K-step)

    const int tid  = threadIdx.x;
    const int lane = tid & 31;
    const int wave = tid >> 5;
    const int wm   = wave >> 1;    // 0..3 -> M-tiles wm and wm+4
    const int wn   = wave & 1;     // 0..1 -> N-tiles 4wn..4wn+3
    const int bm   = blockIdx.y * 128;
    const int bn   = blockIdx.x * 128;

    // Staging assignment: 2 threads per row, 32 contiguous elements each.
    const int srow = tid >> 1;
    const int scol = (tid & 1) * 32;
    const float*  Aptr = A  + (size_t)(bm + srow) * K + scol;
    const __bf16* Wptr = WT + (size_t)(bn + srow) * K + scol;

    float4 areg[8];   // 32 fp32 of A row segment
    uint4  wreg[4];   // 32 bf16 of W row segment

    v8f acc[2][4];    // [m-tile][n-tile]
#pragma unroll
    for (int mi = 0; mi < 2; ++mi)
#pragma unroll
        for (int t = 0; t < 4; ++t)
#pragma unroll
            for (int e = 0; e < 8; ++e) acc[mi][t][e] = 0.0f;

    const int mrow  = lane & 15;
    const int nlane = lane & 15;
    const int akb   = (lane < 16) ? 0 : 8;
    const int bkb   = (lane < 16) ? 0 : 16;

    // prefetch first K-step
#pragma unroll
    for (int i = 0; i < 8; ++i) areg[i] = *(const float4*)(Aptr + 4 * i);
#pragma unroll
    for (int i = 0; i < 4; ++i) wreg[i] = *(const uint4*)(Wptr + 8 * i);

    for (int k0 = 0; k0 < K; k0 += 64) {
        __syncthreads();
        // publish staged tiles: A converted to bf16, W copied, both b128 stores
#pragma unroll
        for (int i = 0; i < 4; ++i) {
            const float* f = (const float*)&areg[2 * i];
            v8bf p;
#pragma unroll
            for (int j = 0; j < 8; ++j) p[j] = (__bf16)f[j];
            *(v8bf*)&As[srow][scol + 8 * i] = p;
        }
#pragma unroll
        for (int i = 0; i < 4; ++i)
            *(uint4*)&Bs[srow][scol + 8 * i] = wreg[i];
        __syncthreads();

        // prefetch next K-step (overlaps the WMMA block below)
        if (k0 + 64 < K) {
#pragma unroll
            for (int i = 0; i < 8; ++i) areg[i] = *(const float4*)(Aptr + k0 + 64 + 4 * i);
#pragma unroll
            for (int i = 0; i < 4; ++i) wreg[i] = *(const uint4*)(Wptr + k0 + 64 + 8 * i);
        }

        // WMMA block: each B fragment feeds two WMMAs (independent accums),
        // so the second WMMA covers the latency of the next fragment load.
#pragma unroll
        for (int s = 0; s < 2; ++s) {
            Frag16 a0, a1;
            a0.h[0] = *(const v8bf*)&As[wm * 16 + mrow][s * 32 + akb];
            a0.h[1] = *(const v8bf*)&As[wm * 16 + mrow][s * 32 + 16 + akb];
            a1.h[0] = *(const v8bf*)&As[(wm + 4) * 16 + mrow][s * 32 + akb];
            a1.h[1] = *(const v8bf*)&As[(wm + 4) * 16 + mrow][s * 32 + 16 + akb];
#pragma unroll
            for (int t = 0; t < 4; ++t) {
                Frag16 b;
                b.h[0] = *(const v8bf*)&Bs[(wn * 4 + t) * 16 + nlane][s * 32 + akb];
                b.h[1] = *(const v8bf*)&Bs[(wn * 4 + t) * 16 + nlane][s * 32 + 16 + akb];
                acc[0][t] = __builtin_amdgcn_wmma_f32_16x16x32_bf16(
                    false, a0.v, false, b.v, (short)0, acc[0][t], false, false);
                acc[1][t] = __builtin_amdgcn_wmma_f32_16x16x32_bf16(
                    false, a1.v, false, b.v, (short)0, acc[1][t], false, false);
            }
        }
    }

    // Epilogue: bias (+residual) (+GELU), fp32 store.
    const int halfm = (lane < 16) ? 0 : 8;
#pragma unroll
    for (int mi = 0; mi < 2; ++mi) {
#pragma unroll
        for (int t = 0; t < 4; ++t) {
            int ncol = bn + (wn * 4 + t) * 16 + nlane;
            float bval = bias[ncol];
#pragma unroll
            for (int r = 0; r < 8; ++r) {
                int mg = bm + (wm + 4 * mi) * 16 + halfm + r;
                float x = acc[mi][t][r] + bval;
                if (res) x += res[(size_t)mg * N + ncol];
                if (act) x = 0.5f * x * (1.0f + erff(x * 0.70710678118654752f));
                C[(size_t)mg * N + ncol] = x;
            }
        }
    }
}

// ---------------------------------------------------------------------------
// Flash attention, one workgroup per (b, head, 128-row Q block).
// 8 waves; wave w owns 16 Q rows. Keys/values processed in tiles of 64.
// Q/K/V/O are (B*S, 768) fp32 with head column offset head*64.
// ---------------------------------------------------------------------------
__global__ __launch_bounds__(256)
void attention_kernel(const float* __restrict__ Qg, const float* __restrict__ Kg,
                      const float* __restrict__ Vg, float* __restrict__ Og)
{
    __shared__ __align__(16) __bf16 Ks[64][TKP];       // keys x d      9 KB
    __shared__ __align__(16) __bf16 Vs[64][TKP];       // d x keys      9 KB
    __shared__ __align__(16) __bf16 Pw[8][16][TKP];    // per-wave P   18 KB

    const int tid  = threadIdx.x;
    const int lane = tid & 31;
    const int wave = tid >> 5;

    int blk = blockIdx.x;
    const int qt   = blk & 3;  blk >>= 2;
    const int head = blk % 12;
    const int b    = blk / 12;
    const int row0 = b * 512 + qt * 128 + wave * 16;
    const int hcol = head * 64;

    const int mrow  = lane & 15;
    const int akb   = (lane < 16) ? 0 : 8;
    const int bkb   = (lane < 16) ? 0 : 16;
    const int nlane = lane & 15;
    const int halfm = (lane < 16) ? 0 : 8;

    // Preload Q fragments (two 32-wide K-steps), bf16 in regs.
    Frag16 aQ[2];
    {
        const float* qrow = Qg + (size_t)(row0 + mrow) * 768 + hcol;
#pragma unroll
        for (int s = 0; s < 2; ++s) {
            aQ[s].h[0] = cvt8(qrow + s * 32 + akb);
            aQ[s].h[1] = cvt8(qrow + s * 32 + 16 + akb);
        }
    }

    float row_max[8], row_sum[8];
#pragma unroll
    for (int r = 0; r < 8; ++r) { row_max[r] = -1e30f; row_sum[r] = 0.0f; }
    v8f acc_o[4];
#pragma unroll
    for (int t = 0; t < 4; ++t)
#pragma unroll
        for (int e = 0; e < 8; ++e) acc_o[t][e] = 0.0f;

    // staging assignment: 4 threads per key row, 16 contiguous d each
    const int krow = tid >> 2;
    const int kcol = (tid & 3) * 16;

    for (int kt = 0; kt < 8; ++kt) {
        const int kv0 = b * 512 + kt * 64;
        __syncthreads();
        {
            const float* kp = Kg + (size_t)(kv0 + krow) * 768 + hcol + kcol;
            *(v8bf*)&Ks[krow][kcol]     = cvt8(kp);
            *(v8bf*)&Ks[krow][kcol + 8] = cvt8(kp + 8);

            const float* vp = Vg + (size_t)(kv0 + krow) * 768 + hcol + kcol;
            float varr[16];
            *(float4*)&varr[0]  = *(const float4*)(vp);
            *(float4*)&varr[4]  = *(const float4*)(vp + 4);
            *(float4*)&varr[8]  = *(const float4*)(vp + 8);
            *(float4*)&varr[12] = *(const float4*)(vp + 12);
#pragma unroll
            for (int i = 0; i < 16; ++i)
                Vs[kcol + i][krow] = (__bf16)varr[i];    // transpose scatter
        }
        __syncthreads();

        // scores tile: Q(16x64) @ K^T(64x64)
        v8f acc_s[4];
#pragma unroll
        for (int t = 0; t < 4; ++t)
#pragma unroll
            for (int e = 0; e < 8; ++e) acc_s[t][e] = 0.0f;
#pragma unroll
        for (int s = 0; s < 2; ++s) {
#pragma unroll
            for (int t = 0; t < 4; ++t) {
                Frag16 bk;
                bk.h[0] = *(const v8bf*)&Ks[t * 16 + nlane][s * 32 + akb];
                bk.h[1] = *(const v8bf*)&Ks[t * 16 + nlane][s * 32 + 16 + akb];
                acc_s[t] = __builtin_amdgcn_wmma_f32_16x16x32_bf16(
                    false, aQ[s].v, false, bk.v, (short)0, acc_s[t], false, false);
            }
        }
#pragma unroll
        for (int t = 0; t < 4; ++t)
#pragma unroll
            for (int e = 0; e < 8; ++e) acc_s[t][e] *= 0.125f;  // 1/sqrt(64)

        // online softmax; xor masks 1..8 reduce over the 16 columns per half.
#pragma unroll
        for (int r = 0; r < 8; ++r) {
            float m = acc_s[0][r];
#pragma unroll
            for (int t = 1; t < 4; ++t) m = fmaxf(m, acc_s[t][r]);
#pragma unroll
            for (int off = 1; off < 16; off <<= 1)
                m = fmaxf(m, __shfl_xor(m, off, 32));
            float mnew = fmaxf(row_max[r], m);
            float corr = __expf(row_max[r] - mnew);
            row_max[r] = mnew;
            float psum = 0.0f;
#pragma unroll
            for (int t = 0; t < 4; ++t) {
                float p = __expf(acc_s[t][r] - mnew);
                acc_s[t][r] = p;
                psum += p;
            }
#pragma unroll
            for (int off = 1; off < 16; off <<= 1)
                psum += __shfl_xor(psum, off, 32);
            row_sum[r] = row_sum[r] * corr + psum;
#pragma unroll
            for (int t = 0; t < 4; ++t) acc_o[t][r] *= corr;
        }

        // spill P (C-layout) to per-wave LDS, re-read in A-fragment layout
#pragma unroll
        for (int t = 0; t < 4; ++t)
#pragma unroll
            for (int r = 0; r < 8; ++r)
                Pw[wave][halfm + r][t * 16 + nlane] = (__bf16)acc_s[t][r];
        __syncthreads();

        // ctx += P(16x64) @ V(64x64)
#pragma unroll
        for (int s = 0; s < 2; ++s) {
            Frag16 aP;
            aP.h[0] = *(const v8bf*)&Pw[wave][mrow][s * 32 + akb];
            aP.h[1] = *(const v8bf*)&Pw[wave][mrow][s * 32 + 16 + akb];
#pragma unroll
            for (int t = 0; t < 4; ++t) {
                Frag16 bv_;
                bv_.h[0] = *(const v8bf*)&Vs[t * 16 + nlane][s * 32 + akb];
                bv_.h[1] = *(const v8bf*)&Vs[t * 16 + nlane][s * 32 + 16 + akb];
                acc_o[t] = __builtin_amdgcn_wmma_f32_16x16x32_bf16(
                    false, aP.v, false, bv_.v, (short)0, acc_o[t], false, false);
            }
        }
    }

    // normalize and write ctx (fp32)
#pragma unroll
    for (int r = 0; r < 8; ++r) {
        float inv = 1.0f / row_sum[r];
        int mg = row0 + halfm + r;
#pragma unroll
        for (int t = 0; t < 4; ++t)
            Og[(size_t)mg * 768 + hcol + t * 16 + nlane] = acc_o[t][r] * inv;
    }
}

// ---------------------------------------------------------------------------
// Row LayerNorm over H=768, one 256-thread block per row. eps = 1e-12.
// Wave-level shfl_xor reductions + 8-entry shared combine.
// ---------------------------------------------------------------------------
__global__ __launch_bounds__(256)
void layernorm_kernel(const float* __restrict__ X, const float* __restrict__ g,
                      const float* __restrict__ bta, float* __restrict__ Y)
{
    __shared__ float red[8];
    const int tid = threadIdx.x, lane = tid & 31, wave = tid >> 5;
    const float* x = X + (size_t)blockIdx.x * 768;

    float vals[3];
    float s = 0.0f;
#pragma unroll
    for (int i = 0; i < 3; ++i) { vals[i] = x[tid + 256 * i]; s += vals[i]; }
#pragma unroll
    for (int off = 1; off < 32; off <<= 1) s += __shfl_xor(s, off, 32);
    if (lane == 0) red[wave] = s;
    __syncthreads();
    float mean = 0.0f;
#pragma unroll
    for (int w = 0; w < 8; ++w) mean += red[w];
    mean *= (1.0f / 768.0f);
    __syncthreads();

    float vs = 0.0f;
#pragma unroll
    for (int i = 0; i < 3; ++i) { float d = vals[i] - mean; vs += d * d; }
#pragma unroll
    for (int off = 1; off < 32; off <<= 1) vs += __shfl_xor(vs, off, 32);
    if (lane == 0) red[wave] = vs;
    __syncthreads();
    float var = 0.0f;
#pragma unroll
    for (int w = 0; w < 8; ++w) var += red[w];
    const float rstd = rsqrtf(var * (1.0f / 768.0f) + 1e-12f);

    float* y = Y + (size_t)blockIdx.x * 768;
#pragma unroll
    for (int i = 0; i < 3; ++i) {
        int c = tid + 256 * i;
        y[c] = (vals[i] - mean) * rstd * g[c] + bta[c];
    }
}

// ---------------------------------------------------------------------------
// Host orchestration
// ---------------------------------------------------------------------------
extern "C" void kernel_launch(void* const* d_in, const int* in_sizes, int n_in,
                              void* d_out, int out_size, void* d_ws, size_t ws_size,
                              hipStream_t stream)
{
    constexpr int Hn = 768, In = 3072, Ln = 6;
    constexpr size_t BS = 8192;          // B*S tokens
    constexpr size_t TOK = BS * Hn;      // 6,291,456 floats

    const float* hidden = (const float*)d_in[0];
    const float* Wq  = (const float*)d_in[1];
    const float* bq  = (const float*)d_in[2];
    const float* Wk  = (const float*)d_in[3];
    const float* bk  = (const float*)d_in[4];
    const float* Wv  = (const float*)d_in[5];
    const float* bv  = (const float*)d_in[6];
    const float* Wo  = (const float*)d_in[7];
    const float* bo  = (const float*)d_in[8];
    const float* g1  = (const float*)d_in[9];
    const float* b1  = (const float*)d_in[10];
    const float* Wi  = (const float*)d_in[11];
    const float* bi  = (const float*)d_in[12];
    const float* Wo2 = (const float*)d_in[13];
    const float* bo2 = (const float*)d_in[14];
    const float* g2  = (const float*)d_in[15];
    const float* b2  = (const float*)d_in[16];

    // Workspace layout (~291 MB)
    float* h     = (float*)d_ws;
    float* q     = h    + TOK;
    float* k     = q    + TOK;
    float* v     = k    + TOK;
    float* ctx   = v    + TOK;
    float* tmp   = ctx  + TOK;
    float* attn  = tmp  + TOK;
    float* inter = attn + TOK;                 // BS * I floats
    __bf16* wq  = (__bf16*)(inter + BS * In);  // all transposed (N x K)
    __bf16* wk  = wq  + (size_t)Hn * Hn;
    __bf16* wv  = wk  + (size_t)Hn * Hn;
    __bf16* wo  = wv  + (size_t)Hn * Hn;
    __bf16* wi  = wo  + (size_t)Hn * Hn;
    __bf16* wo2 = wi  + (size_t)Hn * In;

    auto gemm = [&](const float* A, const __bf16* W, const float* bias,
                    const float* res, float* C, int M, int N, int K, int act) {
        dim3 grid(N / 128, M / 128);
        gemm_bias_kernel<<<grid, 256, 0, stream>>>(A, W, bias, res, C, M, N, K, act);
    };
    auto cvtT = [&](const float* src, __bf16* dst, int K, int N) {
        dim3 grid(N / 32, K / 32);
        convert_transpose_kernel<<<grid, 256, 0, stream>>>(src, dst, K, N);
    };

    hipMemcpyAsync(h, hidden, TOK * sizeof(float), hipMemcpyDeviceToDevice, stream);

    for (int l = 0; l < Ln; ++l) {
        const size_t oHH = (size_t)l * Hn * Hn;
        const size_t oHI = (size_t)l * Hn * In;
        const size_t oH  = (size_t)l * Hn;
        const size_t oI  = (size_t)l * In;

        cvtT(Wq  + oHH, wq,  Hn, Hn);
        cvtT(Wk  + oHH, wk,  Hn, Hn);
        cvtT(Wv  + oHH, wv,  Hn, Hn);
        cvtT(Wo  + oHH, wo,  Hn, Hn);
        cvtT(Wi  + oHI, wi,  Hn, In);
        cvtT(Wo2 + oHI, wo2, In, Hn);

        gemm(h, wq, bq + oH, nullptr, q, (int)BS, Hn, Hn, 0);
        gemm(h, wk, bk + oH, nullptr, k, (int)BS, Hn, Hn, 0);
        gemm(h, wv, bv + oH, nullptr, v, (int)BS, Hn, Hn, 0);

        attention_kernel<<<16 * 12 * 4, 256, 0, stream>>>(q, k, v, ctx);

        gemm(ctx, wo, bo + oH, h, tmp, (int)BS, Hn, Hn, 0);           // + residual h
        layernorm_kernel<<<(int)BS, 256, 0, stream>>>(tmp, g1 + oH, b1 + oH, attn);

        gemm(attn, wi, bi + oI, nullptr, inter, (int)BS, In, Hn, 1);  // exact GELU
        gemm(inter, wo2, bo2 + oH, attn, tmp, (int)BS, Hn, In, 0);    // + residual attn
        layernorm_kernel<<<(int)BS, 256, 0, stream>>>(tmp, g2 + oH, b2 + oH, h);
    }

    hipMemcpyAsync(d_out, h, TOK * sizeof(float), hipMemcpyDeviceToDevice, stream);
}
